// MultiHeadSelfAttention_83932250898756
// MI455X (gfx1250) — matrix operand order repaired and checked by reading, not compiled, not run on hardware
//
#include <hip/hip_runtime.h>

// ---------------------------------------------------------------------------
// Types for CDNA5 WMMA (wave32): bf16 16x16x32, fp32 accumulate
// ---------------------------------------------------------------------------
typedef __attribute__((ext_vector_type(16))) __bf16 v16bf;
typedef __attribute__((ext_vector_type(8)))  float  v8f;
typedef __attribute__((__vector_size__(4 * sizeof(int)))) int int4v;

#define WMMA_BF16(A, B, C) \
    __builtin_amdgcn_wmma_f32_16x16x32_bf16(false, (A), false, (B), (short)0, (C), false, false)

static __device__ __forceinline__ v8f v8f_zero() {
    v8f z;
#pragma unroll
    for (int i = 0; i < 8; ++i) z[i] = 0.0f;
    return z;
}

// A-matrix fragment, 16x32 bf16: lane row = lane&15,
// K = g*16 + 0..15 with g = lane>>4. rowptr -> element [row][k0].
static __device__ __forceinline__ v16bf load_a16x32(const __bf16* __restrict__ rowptr, int g) {
    v16bf a;
    const __bf16* p = rowptr + g * 16;
#pragma unroll
    for (int j = 0; j < 8; ++j) { a[j] = p[j]; a[8 + j] = p[8 + j]; }
    return a;
}

// B-matrix fragment, 32x16 bf16: lane col N = lane&15, K = g*16 + j (j=0..15),
// one contiguous 16-element (32B) run along K. p points at that run.
static __device__ __forceinline__ v16bf load_b32x16(const __bf16* __restrict__ p) {
    v16bf b;
#pragma unroll
    for (int j = 0; j < 16; ++j) b[j] = p[j];
    return b;
}

// ---------------------------------------------------------------------------
// Async global->LDS staging (CDNA5 GLOBAL_LOAD_ASYNC_TO_LDS_B128, ASYNCcnt).
// Builtin signature (from compiler diagnostic): (int4 AS1*, int4 AS3*, imm, imm)
// Guarded: falls back to synchronous 16B copies if builtin not available.
// ---------------------------------------------------------------------------
#if __has_builtin(__builtin_amdgcn_global_load_async_to_lds_b128)
#define USE_ASYNC_LDS 1
#else
#define USE_ASYNC_LDS 0
#endif

static __device__ __forceinline__ void copy16_to_lds(void* lds_dst, const void* g_src) {
#if USE_ASYNC_LDS
    __builtin_amdgcn_global_load_async_to_lds_b128(
        (__attribute__((address_space(1))) int4v*)(g_src),
        (__attribute__((address_space(3))) int4v*)(lds_dst),
        0, 0);
#else
    *(uint4*)lds_dst = *(const uint4*)g_src;
#endif
}

static __device__ __forceinline__ void wait_async_all() {
#if USE_ASYNC_LDS
#if __has_builtin(__builtin_amdgcn_s_wait_asynccnt)
    __builtin_amdgcn_s_wait_asynccnt(0);
#else
    asm volatile("s_wait_asynccnt 0" ::: "memory");
#endif
#endif
}

// ---------------------------------------------------------------------------
// Problem constants (from reference): B=2, S=2048, Dm=1024, H=16, Dh=64
// ---------------------------------------------------------------------------
#define DM   1024
#define SEQ  2048
#define NH   16
#define DH   64
#define MTOT 4096            // B * S
#define NQKV 3072            // 3 * Dm

// Workspace layout (bytes)
#define OFF_XB    (0ull)                 // x bf16        : 8 MB
#define OFF_WQKVB (8ull  << 20)          // Wqkv bf16     : 6 MB
#define OFF_WOUTB (14ull << 20)          // Wout bf16     : 2 MB
#define OFF_QB    (16ull << 20)          // Q (B,H,S,Dh)  : 8 MB
#define OFF_KB    (24ull << 20)          // K (B,H,S,Dh)  : 8 MB
#define OFF_VTB   (32ull << 20)          // V^T (B,H,Dh,S): 8 MB
#define OFF_OB    (40ull << 20)          // attn out (B,S,Dm) bf16: 8 MB

// ---------------------------------------------------------------------------
// Kernel 1: fp32 -> bf16 convert
// ---------------------------------------------------------------------------
__global__ void cvt_f32_to_bf16(const float* __restrict__ s, __bf16* __restrict__ d, int n) {
    int i = blockIdx.x * blockDim.x + threadIdx.x;
    if (i < n) d[i] = (__bf16)s[i];
}

// ---------------------------------------------------------------------------
// Kernel 2: qkv = x @ Wqkv^T + bqkv, then RoPE on q,k; scatter to Q,K,(V^T).
// Block = 256 thr (8 waves), block tile 128(M) x 64(N), wave tile 32x32.
// ---------------------------------------------------------------------------
__global__ __launch_bounds__(256)
void qkv_rope_kernel(const __bf16* __restrict__ X,      // (4096,1024) bf16
                     const __bf16* __restrict__ W,      // (3072,1024) bf16
                     const float*  __restrict__ bias,   // (3072,)
                     __bf16* __restrict__ Q,            // (B,H,S,Dh)
                     __bf16* __restrict__ Kd,           // (B,H,S,Dh)
                     __bf16* __restrict__ Vt)           // (B,H,Dh,S)
{
    const int lane = threadIdx.x & 31;
    const int g    = lane >> 4;
    const int ln   = lane & 15;
    const int w    = threadIdx.x >> 5;
    const int m0   = blockIdx.x * 128 + (w & 3) * 32;
    const int n0   = blockIdx.y * 64  + (w >> 2) * 32;

    v8f acc[2][2];
#pragma unroll
    for (int i = 0; i < 2; ++i)
#pragma unroll
        for (int j = 0; j < 2; ++j) acc[i][j] = v8f_zero();

    for (int k0 = 0; k0 < DM; k0 += 32) {
        v16bf a[2], b[2];
#pragma unroll
        for (int i = 0; i < 2; ++i)
            a[i] = load_a16x32(X + (size_t)(m0 + i * 16 + ln) * DM + k0, g);
#pragma unroll
        for (int j = 0; j < 2; ++j)
            b[j] = load_b32x16(W + (size_t)(n0 + j * 16 + ln) * DM + k0 + g * 16);
        __builtin_prefetch(X + (size_t)(m0 + ln) * DM + k0 + 32, 0, 3);
        __builtin_prefetch(W + (size_t)(n0 + ln) * DM + k0 + 32, 0, 3);
#pragma unroll
        for (int i = 0; i < 2; ++i)
#pragma unroll
            for (int j = 0; j < 2; ++j)
                acc[i][j] = WMMA_BF16(a[i], b[j], acc[i][j]);
    }

    // Epilogue: bias + interleaved partial RoPE + scatter
    // freq(d) = 10000^(-(d&~1)/32) = exp(-(d&~1) * ln(1e4)/32)
    const float kLnTheta = 0.2878231366422116f;   // ln(10000)/32
#pragma unroll
    for (int i = 0; i < 2; ++i) {
#pragma unroll
        for (int j = 0; j < 2; ++j) {
#pragma unroll
            for (int r = 0; r < 8; ++r) {
                const int m = m0 + i * 16 + r + 8 * g;   // C layout: M = r + 8*(lane>=16)
                const int n = n0 + j * 16 + ln;          // C layout: N = lane&15
                float val = acc[i][j][r] + bias[n];
                // neighbor lane holds column n^1 of the same row
                float partner = __shfl_xor(val, 1, 32);
                const int t = n >> 10;          // 0=q 1=k 2=v
                const int rem = n & (DM - 1);
                const int h = rem >> 6;
                const int d = rem & (DH - 1);
                const int s = m & (SEQ - 1);
                const int bb = m >> 11;
                if (t < 2 && d < 32) {
                    float freq = __expf(-(float)(d & ~1) * kLnTheta);
                    float ang  = (float)s * freq;
                    float cs = __cosf(ang);
                    float sn = __sinf(ang);
                    float rot = (d & 1) ? partner : -partner;
                    val = val * cs + rot * sn;
                }
                const __bf16 bv = (__bf16)val;
                const size_t bh = (size_t)(bb * NH + h);
                if (t == 0)      Q [(bh * SEQ + s) * DH + d] = bv;
                else if (t == 1) Kd[(bh * SEQ + s) * DH + d] = bv;
                else             Vt[(bh * DH  + d) * SEQ + s] = bv;
            }
        }
    }
}

// ---------------------------------------------------------------------------
// Kernel 3: flash attention per (b,h). Block = 8 waves x 16 queries = 128 q.
// K/V^T tiles are double-buffer staged into LDS with async global->LDS copies
// (all 8 waves share them); P is staged through LDS for the C->A transpose.
// ---------------------------------------------------------------------------
__global__ __launch_bounds__(256)
void flash_attn_kernel(const __bf16* __restrict__ Q,
                       const __bf16* __restrict__ Kd,
                       const __bf16* __restrict__ Vt,
                       __bf16* __restrict__ Ob)     // (B,S,Dm) bf16
{
    __shared__ __bf16 Klds[2][32 * 64];   // 2 x 4KB : 32 keys x 64 d
    __shared__ __bf16 Vlds[2][64 * 32];   // 2 x 4KB : 64 d x 32 keys (V^T rows)
    __shared__ __bf16 Plds[8][16][36];    // per-wave 16x32 P tile (+pad)

    const int tid  = threadIdx.x;
    const int lane = tid & 31;
    const int g    = lane >> 4;
    const int ln   = lane & 15;
    const int w    = tid >> 5;

    const int bh = blockIdx.y;           // 0..31
    const int bb = bh >> 4;
    const int h  = bh & (NH - 1);
    const int q0 = blockIdx.x * 128 + w * 16;

    const size_t base = (size_t)bh * SEQ * DH;
    const __bf16* Qp  = Q  + base;
    const __bf16* Kp  = Kd + base;
    const __bf16* Vp  = Vt + base;

    // Q fragments: 16 rows x 64 (two K-chunks of 32)
    v16bf aQ[2];
#pragma unroll
    for (int kk = 0; kk < 2; ++kk)
        aQ[kk] = load_a16x32(Qp + (size_t)(q0 + ln) * DH + kk * 32, g);

    float mrun[8], lrun[8];
    v8f o[4];
#pragma unroll
    for (int r = 0; r < 8; ++r) { mrun[r] = -3.0e38f; lrun[r] = 0.0f; }
#pragma unroll
    for (int nt = 0; nt < 4; ++nt) o[nt] = v8f_zero();

    // cooperative staging: 256 threads x 16B = 4KB per tile
    auto stage = [&](int j0, int buf) {
        // K tile: rows j0..j0+31 of (S,Dh) layout are contiguous (4KB)
        const char* ksrc = (const char*)(Kp + (size_t)j0 * DH);
        copy16_to_lds((char*)&Klds[buf][0] + tid * 16, ksrc + tid * 16);
        // V^T tile: 64 rows of 64B, global row stride = SEQ elements
        const int vd = tid >> 2;            // 0..63
        const int vc = (tid & 3) * 16;      // byte offset within 64B row
        const char* vsrc = (const char*)(Vp + (size_t)vd * SEQ + j0) + vc;
        copy16_to_lds((char*)&Vlds[buf][0] + vd * 64 + vc, vsrc);
    };

    const int NCH = SEQ / 32;    // 64 chunks
    stage(0, 0);
    wait_async_all();
    __syncthreads();

    for (int ch = 0; ch < NCH; ++ch) {
        const int buf = ch & 1;
        if (ch + 1 < NCH) stage((ch + 1) * 32, buf ^ 1);

        // scores: 16 queries x 32 keys (two 16-key N tiles), K-dim 64 chained
        v8f s0 = v8f_zero(), s1 = v8f_zero();
#pragma unroll
        for (int kk = 0; kk < 2; ++kk) {
            v16bf b0 = load_b32x16(&Klds[buf][(size_t)ln        * DH + kk * 32 + g * 16]);
            v16bf b1 = load_b32x16(&Klds[buf][(size_t)(16 + ln) * DH + kk * 32 + g * 16]);
            s0 = WMMA_BF16(aQ[kk], b0, s0);
            s1 = WMMA_BF16(aQ[kk], b1, s1);
        }

        // online softmax (scale = 1/sqrt(64) = 0.125); a row lives in one half
#pragma unroll
        for (int r = 0; r < 8; ++r) {
            float x0 = s0[r] * 0.125f;
            float x1 = s1[r] * 0.125f;
            float mx = fmaxf(x0, x1);
#pragma unroll
            for (int off = 8; off >= 1; off >>= 1)
                mx = fmaxf(mx, __shfl_xor(mx, off, 32));
            float mnew = fmaxf(mrun[r], mx);
            float corr = __expf(mrun[r] - mnew);
            float p0 = __expf(x0 - mnew);
            float p1 = __expf(x1 - mnew);
            float rs = p0 + p1;
#pragma unroll
            for (int off = 8; off >= 1; off >>= 1)
                rs += __shfl_xor(rs, off, 32);
            lrun[r] = lrun[r] * corr + rs;
            mrun[r] = mnew;
#pragma unroll
            for (int nt = 0; nt < 4; ++nt) o[nt][r] *= corr;
            // C-layout row = r + 8*g, col = ln (+16 for second tile)
            Plds[w][r + 8 * g][ln]      = (__bf16)p0;
            Plds[w][r + 8 * g][16 + ln] = (__bf16)p1;
        }
        asm volatile("s_wait_dscnt 0" ::: "memory");

        // reload P in A layout (row = ln, K = g*16 + 0..15)
        v16bf pf;
#pragma unroll
        for (int j = 0; j < 8; ++j) {
            pf[j]     = Plds[w][ln][g * 16 + j];
            pf[8 + j] = Plds[w][ln][g * 16 + 8 + j];
        }

        // O += P (16x32keys) x V (32keys x 64d); V^T rows make B-frags contiguous
#pragma unroll
        for (int nt = 0; nt < 4; ++nt) {
            v16bf bv = load_b32x16(&Vlds[buf][(size_t)(nt * 16 + ln) * 32 + g * 16]);
            o[nt] = WMMA_BF16(pf, bv, o[nt]);
        }

        if (ch + 1 < NCH) wait_async_all();   // next chunk's LDS writes done
        __syncthreads();                      // + everyone finished reading buf
    }

    // normalize and scatter to (B,S,Dm)
#pragma unroll
    for (int r = 0; r < 8; ++r) {
        const float inv = 1.0f / lrun[r];
        const int m  = r + 8 * g;
        const int sq = q0 + m;
#pragma unroll
        for (int nt = 0; nt < 4; ++nt) {
            const int d = nt * 16 + ln;
            Ob[((size_t)bb * SEQ + sq) * DM + h * DH + d] = (__bf16)(o[nt][r] * inv);
        }
    }
}

// ---------------------------------------------------------------------------
// Kernel 4: out = Ob @ Wout^T + bout  (fp32 output)
// ---------------------------------------------------------------------------
__global__ __launch_bounds__(256)
void out_proj_kernel(const __bf16* __restrict__ X,     // (4096,1024) bf16
                     const __bf16* __restrict__ W,     // (1024,1024) bf16
                     const float*  __restrict__ bias,  // (1024,)
                     float* __restrict__ out)          // (4096,1024) fp32
{
    const int lane = threadIdx.x & 31;
    const int g    = lane >> 4;
    const int ln   = lane & 15;
    const int w    = threadIdx.x >> 5;
    const int m0   = blockIdx.x * 128 + (w & 3) * 32;
    const int n0   = blockIdx.y * 64  + (w >> 2) * 32;

    v8f acc[2][2];
#pragma unroll
    for (int i = 0; i < 2; ++i)
#pragma unroll
        for (int j = 0; j < 2; ++j) acc[i][j] = v8f_zero();

    for (int k0 = 0; k0 < DM; k0 += 32) {
        v16bf a[2], b[2];
#pragma unroll
        for (int i = 0; i < 2; ++i)
            a[i] = load_a16x32(X + (size_t)(m0 + i * 16 + ln) * DM + k0, g);
#pragma unroll
        for (int j = 0; j < 2; ++j)
            b[j] = load_b32x16(W + (size_t)(n0 + j * 16 + ln) * DM + k0 + g * 16);
        __builtin_prefetch(X + (size_t)(m0 + ln) * DM + k0 + 32, 0, 3);
        __builtin_prefetch(W + (size_t)(n0 + ln) * DM + k0 + 32, 0, 3);
#pragma unroll
        for (int i = 0; i < 2; ++i)
#pragma unroll
            for (int j = 0; j < 2; ++j)
                acc[i][j] = WMMA_BF16(a[i], b[j], acc[i][j]);
    }

#pragma unroll
    for (int i = 0; i < 2; ++i)
#pragma unroll
        for (int j = 0; j < 2; ++j)
#pragma unroll
            for (int r = 0; r < 8; ++r) {
                const int m = m0 + i * 16 + r + 8 * g;
                const int n = n0 + j * 16 + ln;
                out[(size_t)m * DM + n] = acc[i][j][r] + bias[n];
            }
}

// ---------------------------------------------------------------------------
// Host-side launcher
// ---------------------------------------------------------------------------
extern "C" void kernel_launch(void* const* d_in, const int* in_sizes, int n_in,
                              void* d_out, int out_size, void* d_ws, size_t ws_size,
                              hipStream_t stream) {
    const float* x    = (const float*)d_in[0];   // (2,2048,1024)
    const float* Wqkv = (const float*)d_in[1];   // (3072,1024)
    const float* bqkv = (const float*)d_in[2];   // (3072,)
    const float* Wout = (const float*)d_in[3];   // (1024,1024)
    const float* bout = (const float*)d_in[4];   // (1024,)
    float* out = (float*)d_out;

    char* ws = (char*)d_ws;
    __bf16* xb    = (__bf16*)(ws + OFF_XB);
    __bf16* wqkvb = (__bf16*)(ws + OFF_WQKVB);
    __bf16* woutb = (__bf16*)(ws + OFF_WOUTB);
    __bf16* Qb    = (__bf16*)(ws + OFF_QB);
    __bf16* Kb    = (__bf16*)(ws + OFF_KB);
    __bf16* Vtb   = (__bf16*)(ws + OFF_VTB);
    __bf16* Ob    = (__bf16*)(ws + OFF_OB);

    const int nx = MTOT * DM;        // 4194304
    const int nw = NQKV * DM;        // 3145728
    const int no = DM * DM;          // 1048576

    cvt_f32_to_bf16<<<(nx + 255) / 256, 256, 0, stream>>>(x, xb, nx);
    cvt_f32_to_bf16<<<(nw + 255) / 256, 256, 0, stream>>>(Wqkv, wqkvb, nw);
    cvt_f32_to_bf16<<<(no + 255) / 256, 256, 0, stream>>>(Wout, woutb, no);

    // QKV GEMM + RoPE: grid (4096/128, 3072/64)
    qkv_rope_kernel<<<dim3(32, 48), 256, 0, stream>>>(xb, wqkvb, bqkv, Qb, Kb, Vtb);

    // flash attention: grid (2048/128 q-tiles, B*H)
    flash_attn_kernel<<<dim3(16, 32), 256, 0, stream>>>(Qb, Kb, Vtb, Ob);

    // output projection: grid (4096/128, 1024/64)
    out_proj_kernel<<<dim3(32, 16), 256, 0, stream>>>(Ob, woutb, bout, out);
}
